// FilmEncDecLSTM_12867722019351
// MI455X (gfx1250) — compile-verified, matrix operand order checked by reading, not run on hardware
//
#include <hip/hip_runtime.h>
#include <hip/hip_bf16.h>

typedef __attribute__((ext_vector_type(16))) __bf16 v16bf;
typedef __attribute__((ext_vector_type(8)))  float  v8f;

union Frag16 { v16bf v; unsigned u32[8]; };

#define BATCH 512
#define HD    256
#define G4    1024   // 4*Hd
#define TENC  256
#define TDEC  128

__device__ __forceinline__ unsigned short f32_to_bf16(float f) {
  unsigned u = __float_as_uint(f);
  u += 0x7FFFu + ((u >> 16) & 1u);          // round-to-nearest-even
  return (unsigned short)(u >> 16);
}
__device__ __forceinline__ float sigmoidf(float x) { return 1.f / (1.f + __expf(-x)); }

// ---------------------------------------------------------------- utilities
__global__ void zero_kernel(float* p, int n) {
  int i = blockIdx.x * blockDim.x + threadIdx.x;
  if (i < n) p[i] = 0.f;
}

// f32 -> bf16 elementwise (weight conversion + decoder h0 pack)
__global__ void cvt_kernel(const float* __restrict__ src, unsigned short* __restrict__ dst, int n) {
  int i = blockIdx.x * blockDim.x + threadIdx.x;
  if (i < n) dst[i] = f32_to_bf16(src[i]);
}

// ---------------------------------------------------------------- encoder z-init
// z0 = x_hist[:,t,:] @ to_latent_W^T + b ; pack Acat = [bf16(z0), bf16(h0_prev)]
__global__ void zinit_kernel(const float* __restrict__ xh, int t,
                             const float* __restrict__ Wl, const float* __restrict__ bl,
                             const float* __restrict__ h0,
                             float* __restrict__ z, unsigned short* __restrict__ Acat) {
  int idx = blockIdx.x * blockDim.x + threadIdx.x;   // b*HD + h
  int b = idx >> 8, h = idx & 255;
  const float* xr = xh + ((size_t)b * TENC + t) * 32;
  const float* wr = Wl + (size_t)h * 32;
  float s = bl[h];
#pragma unroll
  for (int f = 0; f < 32; ++f) s += xr[f] * wr[f];
  z[idx] = s;
  Acat[(size_t)b * 512 + h]       = f32_to_bf16(s);
  Acat[(size_t)b * 512 + 256 + h] = f32_to_bf16(h0[idx]);
}

// ---------------------------------------------------------------- fused FiLM-LSTM cell
// GEMM (bf16 WMMA, f32 acc) -> GroupNorm(4) -> FiLM -> gates -> c-norm -> h
__global__ __launch_bounds__(256) void cell_kernel(
    const unsigned short* __restrict__ A, int Kdim,          // bf16 (B x Kdim)
    const unsigned short* __restrict__ Wb,                    // bf16 (1024 x Kdim)
    const float* __restrict__ bias,                           // (1024)
    const float* __restrict__ film,                           // (CD x 2048)
    const int*  __restrict__ u_ptr, int u_stride,             // per-sample cond index
    const float* __restrict__ gamma, const float* __restrict__ beta,  // (256)
    float* __restrict__ c_state, float* __restrict__ h_state,         // (B x 256)
    float* __restrict__ z, int z_set,                         // nullable; set vs add
    unsigned short* __restrict__ packZ, int packStride,       // nullable bf16 pack
    const float* __restrict__ h_next)                         // nullable
{
  __shared__ float ldsC[16][G4];                              // 64 KB exactly

  const int tid    = threadIdx.x;
  const int wave   = tid >> 5;
  const int lane   = tid & 31;
  const int laneLo = lane & 15;
  const int hi     = lane >> 4;
  const int rowBase = blockIdx.x * 16;
  const int colBase = wave * 128;

  // ---- GEMM: C[16 x 1024] for this block, wave owns 128 columns
  v8f acc[8];
#pragma unroll
  for (int t = 0; t < 8; ++t)
#pragma unroll
    for (int j = 0; j < 8; ++j) acc[t][j] = 0.f;

  const unsigned short* Arow = A + (size_t)(rowBase + laneLo) * Kdim;
  for (int k0 = 0; k0 < Kdim; k0 += 32) {
    Frag16 a;
#pragma unroll
    for (int p = 0; p < 8; ++p) {
      int k = k0 + ((p & 4) << 2) + hi * 8 + ((p & 3) << 1);
      a.u32[p] = *(const unsigned*)(Arow + k);
    }
#pragma unroll
    for (int t = 0; t < 8; ++t) {
      const unsigned short* Wrow = Wb + (size_t)(colBase + t * 16 + laneLo) * Kdim;
      Frag16 bfr;
#pragma unroll
      for (int p = 0; p < 8; ++p) {
        int k = k0 + ((p & 4) << 2) + hi * 8 + ((p & 3) << 1);
        bfr.u32[p] = *(const unsigned*)(Wrow + k);
      }
      acc[t] = __builtin_amdgcn_wmma_f32_16x16x32_bf16(
          false, a.v, false, bfr.v, (short)0, acc[t], false, false);
    }
  }

  // ---- park C (+bias) in LDS: C layout lane<16: N=lane, M=r ; lane>=16: M=r+8
#pragma unroll
  for (int t = 0; t < 8; ++t) {
    int n = colBase + t * 16 + laneLo;
    float bv = bias[n];
#pragma unroll
    for (int r = 0; r < 8; ++r) ldsC[r + 8 * hi][n] = acc[t][r] + bv;
  }
  __syncthreads();

  // ---- GroupNorm(4): 16 threads per row; thread owns (grp, quarter) of 64 cols
  const int row = tid >> 4;       // 0..15
  const int sub = tid & 15;
  {
    const int grp = sub >> 2, q = sub & 3;
    const float* pC = &ldsC[row][grp * 256 + q * 64];
    float s = 0.f, ss = 0.f;
    for (int i = 0; i < 64; ++i) { float v = pC[i]; s += v; ss += v * v; }
    s  += __shfl_xor(s, 1);  s  += __shfl_xor(s, 2);
    ss += __shfl_xor(ss, 1); ss += __shfl_xor(ss, 2);
    float gmean[4], grstd[4];
    const int lbase = lane & 16;
#pragma unroll
    for (int g = 0; g < 4; ++g) {
      float sg  = __shfl(s,  lbase + g * 4);
      float ssg = __shfl(ss, lbase + g * 4);
      float mu  = sg * (1.f / 256.f);
      float var = ssg * (1.f / 256.f) - mu * mu;
      gmean[g] = mu; grstd[g] = rsqrtf(var + 1e-5f);
    }

    // ---- FiLM + gates: thread owns 16 hidden channels
    const int b   = rowBase + row;
    const int uid = u_ptr[(size_t)b * u_stride];
    const float* fa = film + (size_t)uid * 2048;   // a
    const float* fb = fa + 1024;                    // bb
    float cnew[16], so[16];
    float cs = 0.f, css = 0.f;
#pragma unroll
    for (int i = 0; i < 16; ++i) {
      int h = sub * 16 + i;
      float zi = (ldsC[row][h]       - gmean[0]) * grstd[0];
      float zf = (ldsC[row][256 + h] - gmean[1]) * grstd[1];
      float zo = (ldsC[row][512 + h] - gmean[2]) * grstd[2];
      float zg = (ldsC[row][768 + h] - gmean[3]) * grstd[3];
      zi = zi * (1.f + fa[h])       + fb[h];
      zf = zf * (1.f + fa[256 + h]) + fb[256 + h];
      zo = zo * (1.f + fa[512 + h]) + fb[512 + h];
      zg = zg * (1.f + fa[768 + h]) + fb[768 + h];
      float cold = c_state[(size_t)b * HD + h];
      float cn = sigmoidf(zf) * cold + sigmoidf(zi) * tanhf(zg);
      cnew[i] = cn; so[i] = sigmoidf(zo);
      cs += cn; css += cn * cn;
    }
    // c-norm stats across the 16 row-threads (within one half-wave)
#pragma unroll
    for (int m = 1; m < 16; m <<= 1) { cs += __shfl_xor(cs, m); css += __shfl_xor(css, m); }
    float cmu   = cs * (1.f / 256.f);
    float crstd = rsqrtf(css * (1.f / 256.f) - cmu * cmu + 1e-5f);

#pragma unroll
    for (int i = 0; i < 16; ++i) {
      int h = sub * 16 + i;
      float cn = cnew[i];
      c_state[(size_t)b * HD + h] = cn;
      float cnn = (cn - cmu) * crstd * gamma[h] + beta[h];
      float hn  = so[i] * tanhf(cnn);
      h_state[(size_t)b * HD + h] = hn;
      float zp = hn;
      if (z) {
        float zv = z_set ? hn : (z[(size_t)b * HD + h] + hn);
        z[(size_t)b * HD + h] = zv;
        zp = zv;
      }
      if (packZ) {
        packZ[(size_t)b * packStride + h] = f32_to_bf16(zp);
        if (h_next)
          packZ[(size_t)b * packStride + HD + h] = f32_to_bf16(h_next[(size_t)b * HD + h]);
      }
    }
  }
}

// ---------------------------------------------------------------- multi-head output
// out[b, mm, t, f] = z @ headW^T + head_b ; M=512 rows, N=512 cols, K=256
__global__ __launch_bounds__(256) void head_kernel(
    const unsigned short* __restrict__ zb,   // bf16 (B x 256)
    const unsigned short* __restrict__ Wb,   // bf16 (512 x 256)
    const float* __restrict__ hb,            // (512)
    float* __restrict__ out, int t)
{
  const int wave = threadIdx.x >> 5, lane = threadIdx.x & 31;
  const int laneLo = lane & 15, hi = lane >> 4;
  const int rowBase = blockIdx.x * 16;
  const int col = blockIdx.y * 128 + wave * 16;

  v8f acc;
#pragma unroll
  for (int j = 0; j < 8; ++j) acc[j] = 0.f;

  const unsigned short* Arow = zb + (size_t)(rowBase + laneLo) * HD;
  const unsigned short* Wrow = Wb + (size_t)(col + laneLo) * HD;
  for (int k0 = 0; k0 < HD; k0 += 32) {
    Frag16 a, bfr;
#pragma unroll
    for (int p = 0; p < 8; ++p) {
      int k = k0 + ((p & 4) << 2) + hi * 8 + ((p & 3) << 1);
      a.u32[p]   = *(const unsigned*)(Arow + k);
      bfr.u32[p] = *(const unsigned*)(Wrow + k);
    }
    acc = __builtin_amdgcn_wmma_f32_16x16x32_bf16(
        false, a.v, false, bfr.v, (short)0, acc, false, false);
  }

  const int n = col + laneLo;
  const int mm = n >> 5, f = n & 31;
  const float bv = hb[n];
#pragma unroll
  for (int r = 0; r < 8; ++r) {
    int b = rowBase + r + 8 * hi;
    out[(size_t)b * (16 * TDEC * 32) + (size_t)mm * (TDEC * 32) + (size_t)t * 32 + f] =
        acc[r] + bv;
  }
}

// ---------------------------------------------------------------- host driver
extern "C" void kernel_launch(void* const* d_in, const int* in_sizes, int n_in,
                              void* d_out, int out_size, void* d_ws, size_t ws_size,
                              hipStream_t stream) {
  const float* x_hist      = (const float*)d_in[0];
  const int*   u_hist      = (const int*)  d_in[1];
  const int*   u_horiz     = (const int*)  d_in[2];
  const float* to_latent_W = (const float*)d_in[3];
  const float* to_latent_b = (const float*)d_in[4];
  const float* film_emb    = (const float*)d_in[5];
  const float* enc_W       = (const float*)d_in[6];
  const float* enc_b       = (const float*)d_in[7];
  const float* enc_gamma   = (const float*)d_in[8];
  const float* enc_beta    = (const float*)d_in[9];
  const float* dec0_W      = (const float*)d_in[10];
  const float* dec0_b      = (const float*)d_in[11];
  const float* dec0_gamma  = (const float*)d_in[12];
  const float* dec0_beta   = (const float*)d_in[13];
  const float* dec_W       = (const float*)d_in[14];
  const float* dec_b       = (const float*)d_in[15];
  const float* dec_gamma   = (const float*)d_in[16];
  const float* dec_beta    = (const float*)d_in[17];
  const float* head_W      = (const float*)d_in[18];
  const float* head_b      = (const float*)d_in[19];
  float* out = (float*)d_out;

  char* ws = (char*)d_ws;
  // workspace layout (bytes)
  float* h  = (float*)(ws + 0);                         // [2][B][HD]  1 MB
  float* c  = (float*)(ws + 1048576);                   // [2][B][HD]  1 MB
  float* z  = (float*)(ws + 2097152);                   // [B][HD]     512 KB
  unsigned short* Acat  = (unsigned short*)(ws + 2621440);  // bf16 B x 512
  unsigned short* Acat0 = (unsigned short*)(ws + 3145728);  // bf16 B x 256
  unsigned short* zb    = (unsigned short*)(ws + 3407872);  // bf16 B x 256
  unsigned short* encWb  = (unsigned short*)(ws + 3670016); // 2 x 1024 x 512
  unsigned short* dec0Wb = (unsigned short*)(ws + 5767168); // 1024 x 256
  unsigned short* decWb  = (unsigned short*)(ws + 6291456); // 1024 x 512
  unsigned short* headWb = (unsigned short*)(ws + 7340032); // 512 x 256
  unsigned short* Acat2  = (unsigned short*)(ws + 7602176); // bf16 B x 512

  // zero recurrent state (h and c are contiguous)
  {
    int n = 2 * 2 * BATCH * HD;
    zero_kernel<<<(n + 255) / 256, 256, 0, stream>>>(h, n);
  }
  // weights -> bf16
  cvt_kernel<<<(2 * G4 * 512 + 255) / 256, 256, 0, stream>>>(enc_W,  encWb,  2 * G4 * 512);
  cvt_kernel<<<(G4 * HD + 255) / 256,      256, 0, stream>>>(dec0_W, dec0Wb, G4 * HD);
  cvt_kernel<<<(G4 * 512 + 255) / 256,     256, 0, stream>>>(dec_W,  decWb,  G4 * 512);
  cvt_kernel<<<(512 * HD + 255) / 256,     256, 0, stream>>>(head_W, headWb, 512 * HD);

  float* h0 = h;               float* h1 = h + BATCH * HD;
  float* c0 = c;               float* c1 = c + BATCH * HD;

  // ---------------- encoder scan
  for (int t = 0; t < TENC; ++t) {
    zinit_kernel<<<(BATCH * HD) / 256, 256, 0, stream>>>(
        x_hist, t, to_latent_W, to_latent_b, h0, z, Acat);
    // layer 0: A=[z, h0_prev]; packs [z+h0_new, h1_prev] into Acat2
    cell_kernel<<<BATCH / 16, 256, 0, stream>>>(
        Acat, 512, encWb, enc_b, film_emb, u_hist + t, TENC,
        enc_gamma, enc_beta, c0, h0, z, /*z_set=*/0,
        Acat2, 512, h1);
    // layer 1: no z / pack needed afterwards
    cell_kernel<<<BATCH / 16, 256, 0, stream>>>(
        Acat2, 512, encWb + G4 * 512, enc_b + G4, film_emb, u_hist + t, TENC,
        enc_gamma + HD, enc_beta + HD, c1, h1, nullptr, 0,
        nullptr, 0, nullptr);
  }

  // ---------------- decoder scan
  for (int t = 0; t < TDEC; ++t) {
    // pack bf16(h0_prev) for the K=256 dec0 GEMM
    cvt_kernel<<<(BATCH * HD + 255) / 256, 256, 0, stream>>>(h0, Acat0, BATCH * HD);
    // layer 0 (input = h0 only): z = h0_new ; packs [z, h1_prev] into Acat
    cell_kernel<<<BATCH / 16, 256, 0, stream>>>(
        Acat0, HD, dec0Wb, dec0_b, film_emb, u_horiz + t, TDEC,
        dec0_gamma, dec0_beta, c0, h0, z, /*z_set=*/1,
        Acat, 512, h1);
    // layer 1: z += h1_new ; packs bf16(z) into zb for the head
    cell_kernel<<<BATCH / 16, 256, 0, stream>>>(
        Acat, 512, decWb, dec_b, film_emb, u_horiz + t, TDEC,
        dec_gamma, dec_beta, c1, h1, z, /*z_set=*/0,
        zb, HD, nullptr);
    // multi-head projection -> out[:, :, t, :]
    head_kernel<<<dim3(BATCH / 16, 4), 256, 0, stream>>>(zb, headWb, head_b, out, t);
  }
}
